// Attention_13709535609202
// MI455X (gfx1250) — compile-verified, matrix operand order checked by reading, not run on hardware
//
#include <hip/hip_runtime.h>
#include <hip/hip_bf16.h>
#include <stdint.h>

typedef __attribute__((ext_vector_type(16))) __bf16 v16bf;
typedef __attribute__((ext_vector_type(8)))  __bf16 bf16x8;
typedef __attribute__((ext_vector_type(8)))  float  v8f;

union AFrag { v16bf v; bf16x8 h[2]; };

#define B_ 2
#define C_ 128
#define N_ 4096
#define H_ 4
#define D_ 32

__device__ __forceinline__ v8f wmma_bf16(v16bf a, v16bf b, v8f c) {
  // 8 args: (neg_a, A, neg_b, B, c_mod, C, reuse_a, reuse_b)
  return __builtin_amdgcn_wmma_f32_16x16x32_bf16(false, a, false, b, (short)0, c,
                                                 false, false);
}

__device__ __forceinline__ bf16x8 cvt_row8(const float* p) {
  bf16x8 r;
#pragma unroll
  for (int i = 0; i < 8; ++i) r[i] = (__bf16)p[i];
  return r;
}

// ---------------------------------------------------------------------------
// Kernel 1: QKV projection.  qkv[o,n] = sum_c wqkv[o,c] * x[b,c,n]
// Q -> [b,h,n,d] bf16 (pre-scaled by 1/sqrt(d)); K -> [b,h,n,d]; V -> [b,h,d,n]
// ---------------------------------------------------------------------------
__global__ void qkv_proj_kernel(const float* __restrict__ x,     // [B][C][N]
                                const float* __restrict__ wqkv,  // [384][C]
                                __bf16* __restrict__ Qb,
                                __bf16* __restrict__ Kb,
                                __bf16* __restrict__ Vt) {
  const int lane = threadIdx.x;
  const int lo = lane & 15, hl = lane >> 4;
  const int ntile = blockIdx.x * 16;
  const int otile = blockIdx.y * 16;
  const int b = blockIdx.z;
  const float* xb = x + (size_t)b * C_ * N_;
  const int n = ntile + lo;

  v8f acc = {};
#pragma unroll
  for (int kc = 0; kc < C_; kc += 32) {
    AFrag a;
    const float* wr = wqkv + (size_t)(otile + lo) * C_ + kc;
    a.h[0] = cvt_row8(wr + 8 * hl);
    a.h[1] = cvt_row8(wr + 16 + 8 * hl);
    AFrag bb;
#pragma unroll
    for (int t = 0; t < 8; ++t) {
      bb.h[0][t] = (__bf16)xb[(size_t)(kc + 8 * hl + t) * N_ + n];
      bb.h[1][t] = (__bf16)xb[(size_t)(kc + 16 + 8 * hl + t) * N_ + n];
    }
    acc = wmma_bf16(a.v, bb.v, acc);
  }

  const float scale = 0.17677669529663687f;  // 32^-0.5
#pragma unroll
  for (int r = 0; r < 8; ++r) {
    int o = otile + r + 8 * hl;              // tile never straddles Q/K/V regions
    float val = acc[r];
    if (o < 128) {
      int h = o >> 5, d = o & 31;
      Qb[(((size_t)b * H_ + h) * N_ + n) * D_ + d] = (__bf16)(val * scale);
    } else if (o < 256) {
      int oo = o - 128; int h = oo >> 5, d = oo & 31;
      Kb[(((size_t)b * H_ + h) * N_ + n) * D_ + d] = (__bf16)val;
    } else {
      int oo = o - 256; int h = oo >> 5, d = oo & 31;
      Vt[(((size_t)b * H_ + h) * D_ + d) * N_ + n] = (__bf16)val;
    }
  }
}

// ---------------------------------------------------------------------------
// Kernel 2: flash attention. 256 threads = 8 waves per block; each wave owns a
// 16-row query tile; the block cooperatively stages each 32x32 K/V tile into
// LDS ONCE via GLOBAL_LOAD_ASYNC_TO_LDS_B128 (ASYNCcnt path), giving 8x K/V
// reuse out of LDS instead of 8 redundant L2 streams.
// ---------------------------------------------------------------------------
__global__ void __launch_bounds__(256)
flash_attn_kernel(const __bf16* __restrict__ Qb,
                  const __bf16* __restrict__ Kb,
                  const __bf16* __restrict__ Vt,
                  __bf16* __restrict__ AO) {  // [B][N][128]
  // LDS: [0,2048)   K tile  : 32 rows (tokens)  x 64 B (32 bf16 of d)
  //      [2048,4096) V tile : 32 rows (d)       x 64 B (32 bf16 of j)
  //      [4096 + w*1024)    : per-wave 16x32 bf16 P-transpose scratch
  __shared__ __attribute__((aligned(16))) unsigned char smem[4096 + 8 * 1024];

  const int tid  = threadIdx.x;
  const int w    = tid >> 5;        // wave id 0..7
  const int lane = tid & 31;
  const int lo = lane & 15, hl = lane >> 4;
  const int itile = blockIdx.x * 128 + w * 16;  // 16 query rows per wave
  const int h = blockIdx.y;
  const int b = blockIdx.z;
  const size_t bh = (size_t)b * H_ + h;

  const uint32_t lds_base = (uint32_t)(uintptr_t)(void*)smem;  // addr[31:0] = LDS offset
  const __bf16* kTile = (const __bf16*)(smem);
  const __bf16* vTile = (const __bf16*)(smem + 2048);
  __bf16* lp = (__bf16*)(smem + 4096 + w * 1024);

  AFrag aQ;
  {
    const __bf16* qr = Qb + (bh * N_ + itile + lo) * D_;
    aQ.h[0] = *reinterpret_cast<const bf16x8*>(qr + 8 * hl);
    aQ.h[1] = *reinterpret_cast<const bf16x8*>(qr + 16 + 8 * hl);
  }

  float m[8], l[8];
  v8f acc0 = {}, acc1 = {};
#pragma unroll
  for (int r = 0; r < 8; ++r) { m[r] = -1e30f; l[r] = 0.0f; }

  const __bf16* kbase = Kb + bh * (size_t)N_ * D_;
  const __bf16* vbase = Vt + bh * (size_t)D_ * N_;

  // Per-wave async staging assignment: one b128 per lane per j-step.
  // Waves 0-3 cover the K tile (128 x 16B chunks), waves 4-7 the V tile.
  const int chunk = (w & 3) * 32 + lane;  // 0..127
  const int srow  = chunk >> 2;           // 0..31 (token row for K, d row for V)
  const int scolb = (chunk & 3) * 16;     // byte offset within the 64B row

  for (int j = 0; j < N_; j += 32) {
    // --- stage K/V tiles into LDS (async DMA path, ASYNCcnt) ---
    if (w < 4) {
      uint64_t g = (uint64_t)(uintptr_t)(kbase + (size_t)(j + srow) * D_) + scolb;
      uint32_t lds = lds_base + srow * 64 + scolb;
      asm volatile("global_load_async_to_lds_b128 %0, %1, off"
                   :: "v"(lds), "v"(g) : "memory");
    } else {
      uint64_t g = (uint64_t)(uintptr_t)(vbase + (size_t)srow * N_ + j) + scolb;
      uint32_t lds = lds_base + 2048 + srow * 64 + scolb;
      asm volatile("global_load_async_to_lds_b128 %0, %1, off"
                   :: "v"(lds), "v"(g) : "memory");
    }
    asm volatile("s_wait_asynccnt 0" ::: "memory");
    __syncthreads();  // all tiles visible to all waves

    // --- S = Q * K^T tile (16 x 32), two WMMAs, K-dim = d = 32 ---
    AFrag bK0, bK1;
    {
      const __bf16* kr0 = kTile + (size_t)lo * D_;         // token j+lo
      const __bf16* kr1 = kTile + (size_t)(16 + lo) * D_;  // token j+16+lo
      bK0.h[0] = *reinterpret_cast<const bf16x8*>(kr0 + 8 * hl);
      bK0.h[1] = *reinterpret_cast<const bf16x8*>(kr0 + 16 + 8 * hl);
      bK1.h[0] = *reinterpret_cast<const bf16x8*>(kr1 + 8 * hl);
      bK1.h[1] = *reinterpret_cast<const bf16x8*>(kr1 + 16 + 8 * hl);
    }
    v8f z = {};
    v8f s0 = wmma_bf16(aQ.v, bK0.v, z);
    v8f s1 = wmma_bf16(aQ.v, bK1.v, z);

    // --- online softmax: row max over 16 lanes of each half-wave ---
    float tm[8];
#pragma unroll
    for (int r = 0; r < 8; ++r) tm[r] = fmaxf(s0[r], s1[r]);
#pragma unroll
    for (int off = 1; off < 16; off <<= 1) {
#pragma unroll
      for (int r = 0; r < 8; ++r) tm[r] = fmaxf(tm[r], __shfl_xor(tm[r], off, 32));
    }

    float p0[8], p1[8], rs[8];
#pragma unroll
    for (int r = 0; r < 8; ++r) {
      float mn = fmaxf(m[r], tm[r]);
      float corr = __expf(m[r] - mn);
      m[r] = mn;
      p0[r] = __expf(s0[r] - mn);
      p1[r] = __expf(s1[r] - mn);
      rs[r] = p0[r] + p1[r];
      l[r] *= corr;
      acc0[r] *= corr;
      acc1[r] *= corr;
    }
#pragma unroll
    for (int off = 1; off < 16; off <<= 1) {
#pragma unroll
      for (int r = 0; r < 8; ++r) rs[r] += __shfl_xor(rs[r], off, 32);
    }
#pragma unroll
    for (int r = 0; r < 8; ++r) l[r] += rs[r];

    // --- transpose P (C/D layout) -> A layout via per-wave LDS scratch ---
    // (single-wave region: LDS ops are in-order within a wave)
#pragma unroll
    for (int r = 0; r < 8; ++r) {
      int M = r + 8 * hl;
      lp[M * 32 + lo]      = (__bf16)p0[r];
      lp[M * 32 + 16 + lo] = (__bf16)p1[r];
    }
    AFrag aP;
    aP.h[0] = *reinterpret_cast<const bf16x8*>(&lp[lo * 32 + 8 * hl]);
    aP.h[1] = *reinterpret_cast<const bf16x8*>(&lp[lo * 32 + 16 + 8 * hl]);

    // --- O += P * V  (out tile 16 x 32, two WMMAs over d-halves) ---
    AFrag bV0, bV1;
    {
      const __bf16* vr0 = vTile + (size_t)lo * 32;         // d = lo
      const __bf16* vr1 = vTile + (size_t)(16 + lo) * 32;  // d = 16 + lo
      bV0.h[0] = *reinterpret_cast<const bf16x8*>(vr0 + 8 * hl);
      bV0.h[1] = *reinterpret_cast<const bf16x8*>(vr0 + 16 + 8 * hl);
      bV1.h[0] = *reinterpret_cast<const bf16x8*>(vr1 + 8 * hl);
      bV1.h[1] = *reinterpret_cast<const bf16x8*>(vr1 + 16 + 8 * hl);
    }
    acc0 = wmma_bf16(aP.v, bV0.v, acc0);
    acc1 = wmma_bf16(aP.v, bV1.v, acc1);

    __syncthreads();  // all waves done with K/V tiles before next overwrite
  }

  // --- epilogue: divide by row sum, store [b][n][h*32+d] bf16 ---
#pragma unroll
  for (int r = 0; r < 8; ++r) {
    float inv = 1.0f / l[r];
    int i = itile + r + 8 * hl;
    __bf16* orow = AO + ((size_t)b * N_ + i) * C_ + h * D_;
    orow[lo]      = (__bf16)(acc0[r] * inv);
    orow[16 + lo] = (__bf16)(acc1[r] * inv);
  }
}

// ---------------------------------------------------------------------------
// Kernel 3: output projection + bias.  res[b,o,n] = sum_c wout[o,c]*AO[b,n,c] + bout[o]
// ---------------------------------------------------------------------------
__global__ void out_proj_kernel(const __bf16* __restrict__ AO,   // [B][N][128]
                                const float* __restrict__ wout,  // [128][128]
                                const float* __restrict__ bout,  // [128]
                                float* __restrict__ out) {       // [B][128][N]
  const int lane = threadIdx.x;
  const int lo = lane & 15, hl = lane >> 4;
  const int ntile = blockIdx.x * 16;
  const int otile = blockIdx.y * 16;
  const int b = blockIdx.z;
  const int n = ntile + lo;

  v8f acc = {};
#pragma unroll
  for (int kc = 0; kc < C_; kc += 32) {
    AFrag a;
    const float* wr = wout + (size_t)(otile + lo) * C_ + kc;
    a.h[0] = cvt_row8(wr + 8 * hl);
    a.h[1] = cvt_row8(wr + 16 + 8 * hl);
    AFrag bb;
    const __bf16* ar = AO + ((size_t)b * N_ + n) * C_ + kc;
    bb.h[0] = *reinterpret_cast<const bf16x8*>(ar + 8 * hl);
    bb.h[1] = *reinterpret_cast<const bf16x8*>(ar + 16 + 8 * hl);
    acc = wmma_bf16(a.v, bb.v, acc);
  }
#pragma unroll
  for (int r = 0; r < 8; ++r) {
    int o = otile + r + 8 * hl;
    out[((size_t)b * C_ + o) * N_ + n] = acc[r] + bout[o];
  }
}

// ---------------------------------------------------------------------------
extern "C" void kernel_launch(void* const* d_in, const int* in_sizes, int n_in,
                              void* d_out, int out_size, void* d_ws, size_t ws_size,
                              hipStream_t stream) {
  (void)in_sizes; (void)n_in; (void)out_size; (void)ws_size;
  const float* x    = (const float*)d_in[0];  // [2,128,16,16,16]
  const float* wqkv = (const float*)d_in[1];  // [384,128]
  const float* wout = (const float*)d_in[2];  // [128,128]
  const float* bout = (const float*)d_in[3];  // [128]
  float* out = (float*)d_out;                 // [2,128,16,16,16] f32

  const size_t seg = (size_t)B_ * H_ * N_ * D_;  // 1,048,576 bf16 elements = 2 MB
  __bf16* Qb = (__bf16*)d_ws;
  __bf16* Kb = Qb + seg;
  __bf16* Vt = Kb + seg;
  __bf16* AO = Vt + seg;  // [B][N][128], also 'seg' elements. Total 8 MB.

  qkv_proj_kernel<<<dim3(N_ / 16, 384 / 16, B_), 32, 0, stream>>>(x, wqkv, Qb, Kb, Vt);
  flash_attn_kernel<<<dim3(N_ / 128, H_, B_), 256, 0, stream>>>(Qb, Kb, Vt, AO);
  out_proj_kernel<<<dim3(N_ / 16, C_ / 16, B_), 32, 0, stream>>>(AO, wout, bout, out);
}
